// DPsoftminLoss_3341484556823
// MI455X (gfx1250) — compile-verified, matrix operand order checked by reading, not run on hardware
//
#include <hip/hip_runtime.h>
#include <hip/hip_bf16.h>

typedef __attribute__((ext_vector_type(16))) _Float16     v16h;
typedef __attribute__((ext_vector_type(8)))  float        v8f;
typedef __attribute__((ext_vector_type(8)))  unsigned int v8u;

#define P_PENAL 3.2f
#define TEMP2   0.01f
#define BB 16
#define TT 128
#define JJ 128
#define DD 256

// ---------------------------------------------------------------------------
// Kernel 1: DPLoss[b,t,j] = sum_d |frame[b,t,d] - DPmap[b,t,j,d]|
// One block per (b,t), 8 waves; each wave owns a 16-row j tile and reduces
// D=256 via 8x v_wmma_f32_16x16x32_f16 with an all-ones B matrix.
//
// NOTE: no f32 fabs anywhere — fabsf / fmaxf(x,-x) both canonicalize to
// llvm.fabs on v2f32 after SLP, which this gfx1250 backend fails to select.
// Instead: signed diff -> cvt to f16 (sign-symmetric rounding) -> clear the
// f16 sign bits with an integer AND on the packed dwords (v_and_b32).
// ---------------------------------------------------------------------------
__global__ void __launch_bounds__(256) k_dploss(const float* __restrict__ DPmap,
                                                const float* __restrict__ frame,
                                                float* __restrict__ DPLoss)
{
    const int bt   = blockIdx.x;            // b*128 + t
    const int lane = threadIdx.x & 31;
    const int wave = threadIdx.x >> 5;      // j-tile 0..7
    const int m    = lane & 15;             // row within 16x16 tile
    const int kg   = lane >> 4;             // which K-half this lane holds
    const int j    = (wave << 4) + m;

    const float* __restrict__ dp = DPmap + ((size_t)bt * JJ + j) * DD;
    const float* __restrict__ fr = frame + (size_t)bt * DD;

    v16h ones;
    #pragma unroll
    for (int h = 0; h < 16; ++h) ones[h] = (_Float16)1.0f;

    v8f acc = {};

    #pragma unroll
    for (int dc = 0; dc < DD; dc += 32) {
        // A layout (16-bit A 16x32): lanes 0-15 -> K = 0..7 & 16..23,
        //                            lanes 16-31 -> K = 8..15 & 24..31
        const int d0 = dc + (kg << 3);
        const int d1 = d0 + 16;
        if (dc + 32 < DD) __builtin_prefetch(dp + dc + 32, 0, 0);
        float4 fa = *(const float4*)(fr + d0);
        float4 fb = *(const float4*)(fr + d0 + 4);
        float4 fc = *(const float4*)(fr + d1);
        float4 fd = *(const float4*)(fr + d1 + 4);
        float4 ga = *(const float4*)(dp + d0);
        float4 gb = *(const float4*)(dp + d0 + 4);
        float4 gc = *(const float4*)(dp + d1);
        float4 gd = *(const float4*)(dp + d1 + 4);
        v16h a;
        a[0]  = (_Float16)(fa.x - ga.x); a[1]  = (_Float16)(fa.y - ga.y);
        a[2]  = (_Float16)(fa.z - ga.z); a[3]  = (_Float16)(fa.w - ga.w);
        a[4]  = (_Float16)(fb.x - gb.x); a[5]  = (_Float16)(fb.y - gb.y);
        a[6]  = (_Float16)(fb.z - gb.z); a[7]  = (_Float16)(fb.w - gb.w);
        a[8]  = (_Float16)(fc.x - gc.x); a[9]  = (_Float16)(fc.y - gc.y);
        a[10] = (_Float16)(fc.z - gc.z); a[11] = (_Float16)(fc.w - gc.w);
        a[12] = (_Float16)(fd.x - gd.x); a[13] = (_Float16)(fd.y - gd.y);
        a[14] = (_Float16)(fd.z - gd.z); a[15] = (_Float16)(fd.w - gd.w);
        // |x| in f16: clear the sign bit of each packed half (integer AND).
        v8u au = __builtin_bit_cast(v8u, a);
        au = au & 0x7FFF7FFFu;
        a = __builtin_bit_cast(v16h, au);
        // D[m,n] = sum_k A[m,k] (B = ones); columns replicated.
        acc = __builtin_amdgcn_wmma_f32_16x16x32_f16(false, a, false, ones,
                                                     (short)0, acc, false, false);
    }

    // C/D layout: lane 0 holds rows 0..7 (col 0), lane 16 holds rows 8..15.
    if (m == 0) {
        float* out = DPLoss + (size_t)bt * JJ + (wave << 4) + (kg << 3);
        #pragma unroll
        for (int r = 0; r < 8; ++r) out[r] = acc[r];
    }
}

// ---------------------------------------------------------------------------
// wave32 reductions (2 barriers per block reduction instead of log2(128))
// ---------------------------------------------------------------------------
__device__ __forceinline__ float wave_min(float v) {
    #pragma unroll
    for (int o = 16; o > 0; o >>= 1) v = fminf(v, __shfl_xor(v, o, 32));
    return v;
}
__device__ __forceinline__ float wave_sum(float v) {
    #pragma unroll
    for (int o = 16; o > 0; o >>= 1) v += __shfl_xor(v, o, 32);
    return v;
}
__device__ __forceinline__ float blk_min(float* s4, int j, float v) {
    v = wave_min(v);
    if ((j & 31) == 0) s4[j >> 5] = v;
    __syncthreads();
    float r = fminf(fminf(s4[0], s4[1]), fminf(s4[2], s4[3]));
    __syncthreads();
    return r;
}
__device__ __forceinline__ float blk_sum(float* s4, int j, float v) {
    v = wave_sum(v);
    if ((j & 31) == 0) s4[j >> 5] = v;
    __syncthreads();
    float r = (s4[0] + s4[1]) + (s4[2] + s4[3]);
    __syncthreads();
    return r;
}
__device__ __forceinline__ int blk_argmin(float* s4, int* s4i, int j, float v) {
    int idx = j;
    #pragma unroll
    for (int o = 16; o > 0; o >>= 1) {                  // first-index tie-break
        float ov = __shfl_xor(v, o, 32);
        int   oi = __shfl_xor(idx, o, 32);
        if (ov < v || (ov == v && oi < idx)) { v = ov; idx = oi; }
    }
    if ((j & 31) == 0) { s4[j >> 5] = v; s4i[j >> 5] = idx; }
    __syncthreads();
    float bv = s4[0]; int bi = s4i[0];
    #pragma unroll
    for (int w = 1; w < 4; ++w) {
        float wv = s4[w]; int wi = s4i[w];
        if (wv < bv || (wv == bv && wi < bi)) { bv = wv; bi = wi; }
    }
    __syncthreads();
    return bi;
}

// ---------------------------------------------------------------------------
// Kernel 2: forward DP scan. One block per batch, 128 threads (j = lane).
// prev/prev2 live in registers; cross-thread traffic only via reductions.
// Emits argp[b,t] (= argmin_{j<=t} costMap2 row), per-batch ans partial, and
// lastindexEmb (= argp[b, framelen-1], masks coincide).
// ---------------------------------------------------------------------------
__global__ void __launch_bounds__(128) k_scan(const float* __restrict__ DPLoss,
                                              const int* __restrict__ framelen,
                                              int* __restrict__ argp,
                                              float* __restrict__ ansP,
                                              int* __restrict__ lastidx)
{
    const int b = blockIdx.x;
    const int j = threadIdx.x;
    const float* __restrict__ DL = DPLoss + (size_t)b * TT * JJ;
    const int fl1 = framelen[b] - 1;
    const float INF = __builtin_inff();

    __shared__ float s4[4];
    __shared__ int   s4i[4];

    float pv  = (j == 0) ? DL[0] : 0.0f;   // row0
    float pv2 = pv;
    if (j == 0) {
        argp[b * TT] = 0;
        if (fl1 == 0) { ansP[b] = DL[0]; lastidx[b] = 0; }  // row0 case
    }
    __syncthreads();

    for (int i = 1; i < TT; ++i) {
        const float dl   = DL[i * JJ + j];
        const float dlii = DL[i * JJ + i];
        const bool  lt   = (j < i);

        float mn  = blk_min(s4, j, lt ? pv  : INF);
        float lse = blk_sum(s4, j, lt ? expf(-(pv - mn) * (1.0f / TEMP2)) : 0.0f);
        float mn2 = blk_min(s4, j, lt ? pv2 : INF);

        float nv, nv2;
        if (lt)          { nv  = pv + dl;
                           nv2 = pv2 + dl; }
        else if (j == i) { nv  = -TEMP2 * logf(lse) + P_PENAL + dlii + mn;
                           nv2 = mn2 + P_PENAL + dlii; }
        else             { nv = 0.0f; nv2 = 0.0f; }

        int am = blk_argmin(s4, s4i, j, (j <= i) ? nv2 : INF);
        if (j == 0) {
            argp[b * TT + i] = am;
            if (i == fl1) lastidx[b] = am;
        }
        if (i == fl1) {                       // snapshot: ans contribution
            float mk = blk_min(s4, j, (j <= i) ? nv : INF);
            float ss = blk_sum(s4, j, (j <= i) ? expf(-(nv - mk) * (1.0f / TEMP2)) : 0.0f);
            if (j == 0) ansP[b] = -TEMP2 * logf(ss) + mk;
        }
        pv = nv; pv2 = nv2;
    }
}

// ---------------------------------------------------------------------------
// Kernel 3: backward pass -> switchpos, plus final ans sum.
// row_next recurrence over r = 126..0; diag[d] = savepos[b,d,d].
// ---------------------------------------------------------------------------
__global__ void __launch_bounds__(128) k_backtrack(const int* __restrict__ argp,
                                                   const int* __restrict__ lastidx,
                                                   const int* __restrict__ framelen,
                                                   const float* __restrict__ ansP,
                                                   float* __restrict__ out)
{
    const int b = blockIdx.x;
    const int j = threadIdx.x;
    __shared__ float row[JJ];
    __shared__ float diag[JJ];
    const int fl1 = framelen[b] - 1;
    const int li  = lastidx[b];

    row[j] = (fl1 == TT - 1 && j == li) ? 1.0f : 0.0f;   // save_init row T-1
    if (j == TT - 1) diag[TT - 1] = row[TT - 1];
    __syncthreads();

    for (int r = TT - 2; r >= 0; --r) {
        const float scal = row[r + 1];
        const int   ar   = argp[b * TT + r];
        __syncthreads();
        float add = 0.0f;
        if (r == fl1 && j == li) add += 1.0f;            // save_init row r
        if (j == ar)             add += scal;            // scal * one_hot(argp)
        row[j] += add;
        if (j == r) diag[r] = row[j];
        __syncthreads();
    }

    out[1 + b * JJ + j] = (diag[j] == 1.0f) ? 1.0f : 0.0f;
    if (b == 0 && j == 0) {
        float s = 0.0f;
        for (int q = 0; q < BB; ++q) s += ansP[q];
        out[0] = s;
    }
}

// ---------------------------------------------------------------------------
extern "C" void kernel_launch(void* const* d_in, const int* in_sizes, int n_in,
                              void* d_out, int out_size, void* d_ws, size_t ws_size,
                              hipStream_t stream)
{
    (void)in_sizes; (void)n_in; (void)out_size; (void)ws_size;

    const float* DPmap    = (const float*)d_in[0];
    const float* frame    = (const float*)d_in[1];
    const int*   framelen = (const int*)d_in[2];
    float*       out      = (float*)d_out;

    char* w = (char*)d_ws;
    float* DPLoss  = (float*)w;                                       // 1 MB
    int*   argp    = (int*)  (w + (size_t)BB * TT * JJ * 4);          // 8 KB
    float* ansP    = (float*)(w + (size_t)BB * TT * JJ * 4 + BB * TT * 4);
    int*   lastidx = (int*)  (w + (size_t)BB * TT * JJ * 4 + BB * TT * 4 + 64);

    k_dploss   <<<BB * TT, 256, 0, stream>>>(DPmap, frame, DPLoss);
    k_scan     <<<BB,      128, 0, stream>>>(DPLoss, framelen, argp, ansP, lastidx);
    k_backtrack<<<BB,      128, 0, stream>>>(argp, lastidx, framelen, ansP, out);
}
